// Decoder_42709154791713
// MI455X (gfx1250) — matrix operand order repaired and checked by reading, not compiled, not run on hardware
//
#include <hip/hip_runtime.h>

// ---------------------------------------------------------------------------
// 2-layer LSTM decoder for MI455X (gfx1250, wave32, WMMA bf16 16x16x32).
//   B=64, T=128, D=512, 4D=2048, L=2.
// Strategy:
//   1) Prep: gather+convert embeddings to bf16, convert weights to bf16,
//      fold b_ih+b_hh, copy h/c state into workspace (double-buffered h).
//   2) One big parallel GEMM: G0 = X(8192x512) * W_ih0^T + bias  (bf16 out).
//   3) Serial scan over t: per step two kernels (layer0, layer1), each wave
//      owns a 16(batch)x16(feature) tile and computes all 4 gate tiles with
//      v_wmma_f32_16x16x32_bf16, then does the sigmoid/tanh cell update
//      in-register and writes bf16 h (and f32 output for the top layer).
// ---------------------------------------------------------------------------

typedef __attribute__((ext_vector_type(16))) __bf16 v16bf;
typedef __attribute__((ext_vector_type(8)))  __bf16 v8bf;
typedef __attribute__((ext_vector_type(8)))  float  v8f;

#define DMODEL 512
#define G4     2048   // 4*D
#define NB     64     // batch
#define NT     128    // time steps
#define BT     8192   // NB*NT
#define HL     32768  // NB*DMODEL

__device__ __forceinline__ v8f wmma_bf16(v16bf a, v16bf b, v8f c) {
  return __builtin_amdgcn_wmma_f32_16x16x32_bf16(false, a, false, b,
                                                 (short)0, c, false, false);
}

// A fragment: 16x32 tile of row-major [rows, 512] bf16 matrix.
// ISA layout: lanes 0-15 row=lane hold K {k0..k0+7, k0+16..k0+23},
//             lanes 16-31 row=lane-16 hold K {k0+8..k0+15, k0+24..k0+31}.
__device__ __forceinline__ v16bf load_a_frag(const __bf16* __restrict__ base,
                                             int m0, int k0, int lane) {
  int half = lane >> 4;
  int m = m0 + (lane & 15);
  const __bf16* p = base + (size_t)m * DMODEL + k0 + half * 8;
  v8bf lo = *(const v8bf*)(p);        // global_load_b128
  v8bf hi = *(const v8bf*)(p + 16);   // global_load_b128
  v16bf a;
#pragma unroll
  for (int j = 0; j < 8; ++j) { a[j] = lo[j]; a[j + 8] = hi[j]; }
  return a;
}

// B fragment: 32x16 tile of W^T where W is row-major [2048, 512] bf16.
// B(k,n) = W[n][k] -> contiguous in k. lanes 0-15: n=n0+lane, K={k0..k0+7, k0+16..k0+23};
// lanes 16-31: n=n0+lane-16, K={k0+8..k0+15, k0+24..k0+31}.
__device__ __forceinline__ v16bf load_b_frag(const __bf16* __restrict__ W,
                                             int n0, int k0, int lane) {
  int n = n0 + (lane & 15);
  int kb = k0 + ((lane >> 4) << 3);
  const __bf16* p = W + (size_t)n * DMODEL + kb;
  v8bf lo = *(const v8bf*)(p);
  v8bf hi = *(const v8bf*)(p + 16);
  v16bf b;
#pragma unroll
  for (int j = 0; j < 8; ++j) { b[j] = lo[j]; b[j + 8] = hi[j]; }
  return b;
}

__device__ __forceinline__ float sigmoidf_fast(float x) {
  return 1.0f / (1.0f + __expf(-x));
}

// ------------------------------ prep kernels -------------------------------

__global__ void k_cvt_f32_bf16(const float* __restrict__ src,
                               __bf16* __restrict__ dst, int n) {
  for (int i = blockIdx.x * blockDim.x + threadIdx.x; i < n;
       i += gridDim.x * blockDim.x)
    dst[i] = (__bf16)src[i];
}

__global__ void k_embed_gather(const int* __restrict__ tok,   // [BT]
                               const float* __restrict__ emb, // [VOCAB, D]
                               __bf16* __restrict__ Xb) {     // [BT, D]
  int i = blockIdx.x * blockDim.x + threadIdx.x;              // over BT*D
  if (i >= BT * DMODEL) return;
  int row = i >> 9, col = i & (DMODEL - 1);
  Xb[i] = (__bf16)emb[(size_t)tok[row] * DMODEL + col];
}

__global__ void k_bias_sum(const float* __restrict__ b_ih,
                           const float* __restrict__ b_hh,
                           float* __restrict__ bsum, int n) {
  int i = blockIdx.x * blockDim.x + threadIdx.x;
  if (i < n) bsum[i] = b_ih[i] + b_hh[i];
}

__global__ void k_init_state(const float* __restrict__ hiddens, // [2,NB,D]
                             const float* __restrict__ cells,   // [2,NB,D]
                             __bf16* __restrict__ hbuf,         // [2 layer][2 parity][HL]
                             float* __restrict__ cws) {         // [2,HL]
  int i = blockIdx.x * blockDim.x + threadIdx.x;                // over 2*HL
  if (i >= 2 * HL) return;
  int l = i >> 15, idx = i & (HL - 1);
  hbuf[(size_t)l * 2 * HL + idx] = (__bf16)hiddens[i];          // parity 0
  cws[i] = cells[i];
}

// -------- hoisted input projection: G0 = X * W_ih0^T + (b_ih0+b_hh0) -------

__global__ void k_gemm_ih0(const __bf16* __restrict__ Xb,    // [BT, 512]
                           const __bf16* __restrict__ Wih0,  // [2048, 512]
                           const float* __restrict__ bsum0,  // [2048]
                           __bf16* __restrict__ G0) {        // [BT, 2048]
  int wave = (blockIdx.x * blockDim.x + threadIdx.x) >> 5;   // 65536 waves
  int lane = threadIdx.x & 31;
  int rt = wave >> 7;            // 0..511 row tiles
  int nt = wave & 127;           // 0..127 col tiles
  int m0 = rt * 16, n0 = nt * 16;
  int nlo = lane & 15, hhalf = (lane >> 4) * 8;

  float bias = bsum0[n0 + nlo];
  v8f acc;
#pragma unroll
  for (int r = 0; r < 8; ++r) acc[r] = bias;

#pragma unroll
  for (int kc = 0; kc < 16; ++kc) {
    int k0 = kc * 32;
    v16bf a = load_a_frag(Xb, m0, k0, lane);
    v16bf b = load_b_frag(Wih0, n0, k0, lane);
    acc = wmma_bf16(a, b, acc);
  }
#pragma unroll
  for (int r = 0; r < 8; ++r) {
    int row = m0 + r + hhalf;
    G0[(size_t)row * G4 + n0 + nlo] = (__bf16)acc[r];
  }
}

// ------------------------- recurrent step kernels --------------------------
// 128 waves: mt=wave>>5 (batch tile 0..3), dt=wave&31 (feature tile 0..31).
// Each wave computes the 4 gate tiles {i,f,g,o} for its (batch,feature) tile.

__global__ void k_lstm_step_l0(const __bf16* __restrict__ G0,   // [BT, 2048]
                               const __bf16* __restrict__ Whh,  // [2048, 512]
                               const __bf16* __restrict__ h_in, // [NB, 512]
                               __bf16* __restrict__ h_out,      // [NB, 512]
                               float* __restrict__ c,           // [NB, 512]
                               int t) {
  int wave = (blockIdx.x * blockDim.x + threadIdx.x) >> 5;
  int lane = threadIdx.x & 31;
  int m0 = (wave >> 5) * 16;
  int d0 = (wave & 31) * 16;
  int nlo = lane & 15, hhalf = (lane >> 4) * 8;

  v8f acc[4];
#pragma unroll
  for (int g = 0; g < 4; ++g)
#pragma unroll
    for (int r = 0; r < 8; ++r) {
      int b = m0 + r + hhalf;
      acc[g][r] = (float)G0[((size_t)b * NT + t) * G4 + g * DMODEL + d0 + nlo];
    }

#pragma unroll
  for (int kc = 0; kc < 16; ++kc) {
    int k0 = kc * 32;
    v16bf a = load_a_frag(h_in, m0, k0, lane);
#pragma unroll
    for (int g = 0; g < 4; ++g) {
      v16bf b = load_b_frag(Whh, g * DMODEL + d0, k0, lane);
      acc[g] = wmma_bf16(a, b, acc[g]);
    }
  }

#pragma unroll
  for (int r = 0; r < 8; ++r) {
    int b = m0 + r + hhalf;
    int d = d0 + nlo;
    float ig = sigmoidf_fast(acc[0][r]);
    float fg = sigmoidf_fast(acc[1][r]);
    float gg = tanhf(acc[2][r]);
    float og = sigmoidf_fast(acc[3][r]);
    size_t idx = (size_t)b * DMODEL + d;
    float cn = fg * c[idx] + ig * gg;
    c[idx] = cn;
    h_out[idx] = (__bf16)(og * tanhf(cn));
  }
}

__global__ void k_lstm_step_l1(const __bf16* __restrict__ Wih,  // [2048, 512]
                               const __bf16* __restrict__ Whh,  // [2048, 512]
                               const float* __restrict__ bsum,  // [2048]
                               const __bf16* __restrict__ x_in, // h0_new [NB,512]
                               const __bf16* __restrict__ h_in, // [NB, 512]
                               __bf16* __restrict__ h_out,      // [NB, 512]
                               float* __restrict__ c,           // [NB, 512]
                               float* __restrict__ out,         // [NB, NT, 512]
                               int t) {
  int wave = (blockIdx.x * blockDim.x + threadIdx.x) >> 5;
  int lane = threadIdx.x & 31;
  int m0 = (wave >> 5) * 16;
  int d0 = (wave & 31) * 16;
  int nlo = lane & 15, hhalf = (lane >> 4) * 8;

  v8f acc[4];
#pragma unroll
  for (int g = 0; g < 4; ++g) {
    float bias = bsum[g * DMODEL + d0 + nlo];
#pragma unroll
    for (int r = 0; r < 8; ++r) acc[g][r] = bias;
  }

#pragma unroll
  for (int kc = 0; kc < 16; ++kc) {
    int k0 = kc * 32;
    v16bf ax = load_a_frag(x_in, m0, k0, lane);
    v16bf ah = load_a_frag(h_in, m0, k0, lane);
#pragma unroll
    for (int g = 0; g < 4; ++g) {
      v16bf bi = load_b_frag(Wih, g * DMODEL + d0, k0, lane);
      acc[g] = wmma_bf16(ax, bi, acc[g]);
      v16bf bh = load_b_frag(Whh, g * DMODEL + d0, k0, lane);
      acc[g] = wmma_bf16(ah, bh, acc[g]);
    }
  }

#pragma unroll
  for (int r = 0; r < 8; ++r) {
    int b = m0 + r + hhalf;
    int d = d0 + nlo;
    float ig = sigmoidf_fast(acc[0][r]);
    float fg = sigmoidf_fast(acc[1][r]);
    float gg = tanhf(acc[2][r]);
    float og = sigmoidf_fast(acc[3][r]);
    size_t idx = (size_t)b * DMODEL + d;
    float cn = fg * c[idx] + ig * gg;
    c[idx] = cn;
    float hn = og * tanhf(cn);
    h_out[idx] = (__bf16)hn;
    out[((size_t)b * NT + t) * DMODEL + d] = hn;   // top-layer hidden
  }
}

// ------------------------------- launcher ----------------------------------

extern "C" void kernel_launch(void* const* d_in, const int* in_sizes, int n_in,
                              void* d_out, int out_size, void* d_ws, size_t ws_size,
                              hipStream_t stream) {
  const int*   tok     = (const int*)d_in[0];    // [NB, NT] int32
  // d_in[1] encoder_out, d_in[4] src_mask: unused by the reference
  const float* hiddens = (const float*)d_in[2];  // [2, NB, D]
  const float* cells   = (const float*)d_in[3];  // [2, NB, D]
  const float* emb     = (const float*)d_in[5];  // [VOCAB, D]
  const float* W_ih    = (const float*)d_in[6];  // [2, 2048, 512]
  const float* W_hh    = (const float*)d_in[7];  // [2, 2048, 512]
  const float* b_ih    = (const float*)d_in[8];  // [2, 2048]
  const float* b_hh    = (const float*)d_in[9];  // [2, 2048]
  float* out = (float*)d_out;                    // [NB, NT, D]

  // Workspace carve (all offsets 256B-aligned).
  char* ws = (char*)d_ws;
  __bf16* Xb   = (__bf16*)ws;  ws += (size_t)BT * DMODEL * 2;       //  8 MB
  __bf16* Wihb = (__bf16*)ws;  ws += (size_t)2 * G4 * DMODEL * 2;   //  4 MB
  __bf16* Whhb = (__bf16*)ws;  ws += (size_t)2 * G4 * DMODEL * 2;   //  4 MB
  __bf16* G0   = (__bf16*)ws;  ws += (size_t)BT * G4 * 2;           // 32 MB
  float*  bsum = (float*)ws;   ws += (size_t)2 * G4 * 4;
  __bf16* hbuf = (__bf16*)ws;  ws += (size_t)2 * 2 * HL * 2;        // [l][parity][HL]
  float*  cws  = (float*)ws;   ws += (size_t)2 * HL * 4;

  const int nW = 2 * G4 * DMODEL;  // 2,097,152
  k_cvt_f32_bf16<<<nW / 256, 256, 0, stream>>>(W_ih, Wihb, nW);
  k_cvt_f32_bf16<<<nW / 256, 256, 0, stream>>>(W_hh, Whhb, nW);
  k_embed_gather<<<(BT * DMODEL) / 256, 256, 0, stream>>>(tok, emb, Xb);
  k_bias_sum<<<(2 * G4) / 256, 256, 0, stream>>>(b_ih, b_hh, bsum, 2 * G4);
  k_init_state<<<(2 * HL) / 256, 256, 0, stream>>>(hiddens, cells, hbuf, cws);

  // Hoisted input projection over all B*T rows: 65536 wave-tiles.
  k_gemm_ih0<<<8192, 256, 0, stream>>>(Xb, Wihb, bsum, G0);

  // Serial scan: 128 steps x 2 layers; 128 waves (16 blocks x 8 waves) each.
  for (int t = 0; t < NT; ++t) {
    int p = t & 1;
    const __bf16* h0_in = hbuf + (size_t)p * HL;
    __bf16* h0_out      = hbuf + (size_t)(1 - p) * HL;
    const __bf16* h1_in = hbuf + (size_t)2 * HL + (size_t)p * HL;
    __bf16* h1_out      = hbuf + (size_t)2 * HL + (size_t)(1 - p) * HL;

    k_lstm_step_l0<<<16, 256, 0, stream>>>(G0, Whhb, h0_in, h0_out, cws, t);
    k_lstm_step_l1<<<16, 256, 0, stream>>>(Wihb + (size_t)G4 * DMODEL,
                                           Whhb + (size_t)G4 * DMODEL,
                                           bsum + G4, h0_out, h1_in, h1_out,
                                           cws + HL, out, t);
  }
}
